// pam_56049323213661
// MI455X (gfx1250) — matrix operand order repaired and checked by reading, not compiled, not run on hardware
//
#include <hip/hip_runtime.h>

typedef __attribute__((ext_vector_type(16))) _Float16 v16h;
typedef __attribute__((ext_vector_type(8)))  _Float16 v8h;
typedef __attribute__((ext_vector_type(8)))  float    v8f;
typedef __attribute__((ext_vector_type(4)))  float    v4f;

#define WMMA_F16(A,B,C) __builtin_amdgcn_wmma_f32_16x16x32_f16(false,(A),false,(B),(short)0,(C),false,false)

constexpr int Bn  = 8;
constexpr int HWn = 4096;
constexpr int Cn  = 512;
constexpr int Dn  = 64;
constexpr int NTOT  = 640;   // 64 (q) + 64 (k) + 512 (v) output columns
constexpr int KSTEP = 128;   // keys processed per flash iteration

// ---------------------------------------------------------------------------
// Kernel 0: pack [Wq|Wk|Wv] into transposed f16 WT[n][kk] so WMMA B-fragments
// are contiguous 32B loads per lane.
// ---------------------------------------------------------------------------
__global__ void pack_weights(const float* __restrict__ Wq,
                             const float* __restrict__ Wk,
                             const float* __restrict__ Wv,
                             _Float16* __restrict__ WT) {
    int i = blockIdx.x * blockDim.x + threadIdx.x;
    if (i >= NTOT * Cn) return;
    int n  = i / Cn;
    int kk = i - n * Cn;
    float w;
    if (n < 64)       w = Wq[kk * 64 + n];
    else if (n < 128) w = Wk[kk * 64 + (n - 64)];
    else              w = Wv[kk * Cn + (n - 128)];
    WT[(size_t)n * Cn + kk] = (_Float16)w;
}

// ---------------------------------------------------------------------------
// Kernel 1: fused QKV projection GEMM. One wave -> one 16x16 output tile.
// A = x (f32 -> f16 on the fly), B = WT (f16). q,k stored row-major f16;
// v stored TRANSPOSED: vT[b][c][pix] f16 (so P@V B-frags are contiguous).
// ---------------------------------------------------------------------------
__global__ __launch_bounds__(128) void qkv_gemm(
    const float* __restrict__ x, const _Float16* __restrict__ WT,
    const float* __restrict__ bq, const float* __restrict__ bk,
    const float* __restrict__ bv,
    _Float16* __restrict__ qout, _Float16* __restrict__ kout,
    _Float16* __restrict__ vT)
{
    const int lane = threadIdx.x & 31;
    const int nl = lane & 15;
    const int g  = lane >> 4;
    const int wid = blockIdx.x * 4 + (threadIdx.x >> 5);
    const int rt = wid / 40;            // row tile (16 pixels)
    const int nt = wid - rt * 40;       // output-column tile
    const int row0 = rt * 16;
    const int n0   = nt * 16;

    const float*    xrow = x  + (size_t)(row0 + nl) * Cn;   // A row m = nl
    const _Float16* wrow = WT + (size_t)(n0  + nl) * Cn;    // B col n = nl

    v8f acc = {};
    for (int kbs = 0; kbs < Cn; kbs += 32) {
        // A-frag: half j -> d = kbs + 16g + j (contiguous)
        v4f a0 = *(const v4f*)(xrow + kbs + 16*g);
        v4f a1 = *(const v4f*)(xrow + kbs + 16*g + 4);
        v4f a2 = *(const v4f*)(xrow + kbs + 16*g + 8);
        v4f a3 = *(const v4f*)(xrow + kbs + 16*g + 12);
        v16h A;
        #pragma unroll
        for (int j = 0; j < 4; ++j) {
            A[j]      = (_Float16)a0[j];
            A[j + 4]  = (_Float16)a1[j];
            A[j + 8]  = (_Float16)a2[j];
            A[j + 12] = (_Float16)a3[j];
        }
        // B-frag: half j -> K = kbs + 16g + j (contiguous 32B)
        v16h Bf = *(const v16h*)(wrow + kbs + 16*g);
        acc = WMMA_F16(A, Bf, acc);
    }

    const int nglob = n0 + nl;
    float bias;
    if (nglob < 64)       bias = bq[nglob];
    else if (nglob < 128) bias = bk[nglob - 64];
    else                  bias = bv[nglob - 128];
    #pragma unroll
    for (int r = 0; r < 8; ++r) acc[r] += bias;

    if (nt < 8) {                       // q / k region, row-major f16
        _Float16* dst = (nglob < 64) ? qout : kout;
        const int col = nglob & 63;
        #pragma unroll
        for (int r = 0; r < 8; ++r)     // D: vgpr r -> row m = r + 8g
            dst[(size_t)(row0 + r + 8*g) * Dn + col] = (_Float16)acc[r];
    } else {                            // v region, transposed [b][c][pix]
        const int c   = nglob - 128;
        const int b   = row0 >> 12;
        const int pix = (row0 & (HWn - 1)) + 8*g;
        v8h pk;
        #pragma unroll
        for (int r = 0; r < 8; ++r) pk[r] = (_Float16)acc[r];
        *(v8h*)(vT + ((size_t)(b * Cn + c) * HWn + pix)) = pk;
    }
}

// ---------------------------------------------------------------------------
// Kernel 2: flash attention, 128-key steps. 1 block = 4 waves = 16 query rows.
// Phase S : each wave computes S = q k^T for its own 32-key slice (4 WMMAs).
// Phase SM: online softmax fully parallel over 128 threads (row x 16-col seg);
//           per-segment running sums kept in registers (linear recurrence).
// Phase PV: each wave: one alpha-rescale then 32 WMMAs over its 128 channels,
//           V B-frags streamed straight from L2-resident transposed vT.
// ---------------------------------------------------------------------------
__global__ __launch_bounds__(128) void flash_attn(
    const float* __restrict__ x, const float* __restrict__ gamma,
    const _Float16* __restrict__ q, const _Float16* __restrict__ k,
    const _Float16* __restrict__ vT, float* __restrict__ out)
{
    __shared__ __align__(16) float    sS[16][132];   // pitch 132: bank-offset rows
    __shared__ __align__(16) _Float16 sP[16][136];   // pitch 136: 16B-aligned rows
    __shared__ float sPmax[16][8];
    __shared__ float sAlpha[16];
    __shared__ float sM[16];
    __shared__ float sLseg[16][8];

    const int tid  = threadIdx.x;
    const int lane = tid & 31;
    const int nl = lane & 15;
    const int g  = lane >> 4;
    const int wave = tid >> 5;

    const int sm   = tid & 15;          // stats: row owned by this thread
    const int sseg = tid >> 4;          // stats: 16-column segment (0..7)

    const int tile = blockIdx.x;        // 2048 tiles = 8 batches * 256
    const int b  = tile >> 8;
    const int m0 = (tile & 255) << 4;   // query pixel base in batch

    if (tid < 16) sM[tid] = -3.0e38f;
    float lseg = 0.0f;
    __syncthreads();

    // q A-frags for d=0..31 and d=32..63 (row m = nl) -- needed by every wave
    const _Float16* qrow = q + (size_t)(b * HWn + m0 + nl) * Dn;
    v16h qa[2];
    #pragma unroll
    for (int c = 0; c < 2; ++c) {
        v8h lo = *(const v8h*)(qrow + 32*c + 16*g);
        v8h hi = *(const v8h*)(qrow + 32*c + 16*g + 8);
        #pragma unroll
        for (int j = 0; j < 8; ++j) { qa[c][j] = lo[j]; qa[c][j + 8] = hi[j]; }
    }

    v8f acc[8] = {};                    // 8 N-tiles x 16 rows -> 64 VGPRs
    const int c0 = wave * 128;          // this wave's channel slice
    const _Float16* vbase = vT + (size_t)(b * Cn + c0 + nl) * HWn;

    for (int kbs = 0; kbs < HWn; kbs += KSTEP) {
        // ---- Phase S: this wave's 32-key slice of S = q k^T ----
        const int kw = kbs + wave * 32;
        #pragma unroll
        for (int ks = 0; ks < 2; ++ks) {            // two 16-key subtiles
            const _Float16* kp =
                k + (size_t)(b * HWn + kw + ks*16 + nl) * Dn + 16*g;
            v16h kb0 = *(const v16h*)kp;            // d = 16g+j
            v16h kb1 = *(const v16h*)(kp + 32);     // d = 32+16g+j
            v8f s = {};
            s = WMMA_F16(qa[0], kb0, s);
            s = WMMA_F16(qa[1], kb1, s);
            #pragma unroll
            for (int r = 0; r < 8; ++r)             // m = r+8g, n = nl
                sS[r + 8*g][wave*32 + ks*16 + nl] = s[r];
        }
        __syncthreads();

        // ---- Phase SM-a: partial max over own 16 columns ----
        float mold = sM[sm];            // read BEFORE sM is rewritten
        v4f sv[4];
        #pragma unroll
        for (int u = 0; u < 4; ++u)
            sv[u] = *(const v4f*)&sS[sm][sseg*16 + 4*u];
        float pmax = sv[0][0];
        #pragma unroll
        for (int u = 0; u < 4; ++u)
            #pragma unroll
            for (int j = 0; j < 4; ++j) pmax = fmaxf(pmax, sv[u][j]);
        sPmax[sm][sseg] = pmax;
        __syncthreads();

        // ---- Phase SM-b: row max, alpha, exp, emit P ----
        float mnew = mold;
        #pragma unroll
        for (int u = 0; u < 8; ++u) mnew = fmaxf(mnew, sPmax[sm][u]);
        float alpha = __expf(mold - mnew);
        v8h p0, p1;
        float ssum = 0.0f;
        #pragma unroll
        for (int u = 0; u < 4; ++u) {
            #pragma unroll
            for (int j = 0; j < 4; ++j) {
                float p = __expf(sv[u][j] - mnew);
                ssum += p;
                if (u < 2) p0[u*4 + j] = (_Float16)p;
                else       p1[(u-2)*4 + j] = (_Float16)p;
            }
        }
        lseg = lseg * alpha + ssum;     // linear recurrence, summed at end
        *(v8h*)&sP[sm][sseg*16]     = p0;
        *(v8h*)&sP[sm][sseg*16 + 8] = p1;
        if (sseg == 0) { sM[sm] = mnew; sAlpha[sm] = alpha; }
        __syncthreads();

        // ---- Phase PV: one rescale, then 4 x 8 WMMAs ----
        float al[8];
        #pragma unroll
        for (int r = 0; r < 8; ++r) al[r] = sAlpha[r + 8*g];
        #pragma unroll
        for (int t = 0; t < 8; ++t)
            #pragma unroll
            for (int r = 0; r < 8; ++r) acc[t][r] *= al[r];

        #pragma unroll
        for (int s = 0; s < 4; ++s) {   // four 32-key sub-blocks
            // P A-frag: half j -> key s*32+16g+j (contiguous)
            v8h plo = *(const v8h*)&sP[nl][s*32 + 16*g];
            v8h phi = *(const v8h*)&sP[nl][s*32 + 16*g + 8];
            v16h pa;
            #pragma unroll
            for (int j = 0; j < 8; ++j) { pa[j] = plo[j]; pa[j + 8] = phi[j]; }
            #pragma unroll
            for (int t = 0; t < 8; ++t) {
                const _Float16* vp =
                    vbase + (size_t)t * 16 * HWn + kbs + s*32 + 16*g;
                v16h vb = *(const v16h*)vp;   // key = kbs+s*32+16g+j
                __builtin_prefetch(vp + KSTEP, 0, 0);
                acc[t] = WMMA_F16(pa, vb, acc[t]);
            }
        }
    }

    // ---- epilogue: l = sum of per-segment sums, then gamma/l + residual ----
    sLseg[sm][sseg] = lseg;
    __syncthreads();

    const float gm = gamma[0];
    float linv[8];
    #pragma unroll
    for (int r = 0; r < 8; ++r) {
        float l = 0.0f;
        #pragma unroll
        for (int u = 0; u < 8; ++u) l += sLseg[r + 8*g][u];
        linv[r] = 1.0f / l;
    }

    #pragma unroll
    for (int t = 0; t < 8; ++t) {
        size_t base = (size_t)(b * HWn + m0 + 8*g) * Cn + c0 + t*16 + nl;
        #pragma unroll
        for (int r = 0; r < 8; ++r) {
            size_t idx = base + (size_t)r * Cn;
            out[idx] = gm * (acc[t][r] * linv[r]) + x[idx];
        }
    }
}

// ---------------------------------------------------------------------------
extern "C" void kernel_launch(void* const* d_in, const int* in_sizes, int n_in,
                              void* d_out, int out_size, void* d_ws, size_t ws_size,
                              hipStream_t stream)
{
    const float* x     = (const float*)d_in[0];
    const float* Wq    = (const float*)d_in[1];
    const float* bq    = (const float*)d_in[2];
    const float* Wk    = (const float*)d_in[3];
    const float* bk    = (const float*)d_in[4];
    const float* Wv    = (const float*)d_in[5];
    const float* bv    = (const float*)d_in[6];
    const float* gamma = (const float*)d_in[7];
    float* out = (float*)d_out;

    char* ws = (char*)d_ws;
    _Float16* WT   = (_Float16*)(ws);                               // 0.63 MB
    _Float16* qbuf = (_Float16*)(ws + (size_t)(1)  * (1u << 20));   // 4 MB
    _Float16* kbuf = (_Float16*)(ws + (size_t)(6)  * (1u << 20));   // 4 MB
    _Float16* vTb  = (_Float16*)(ws + (size_t)(11) * (1u << 20));   // 32 MB

    pack_weights<<<(NTOT * Cn + 255) / 256, 256, 0, stream>>>(Wq, Wk, Wv, WT);
    qkv_gemm<<<(2048 * 40) / 4, 128, 0, stream>>>(x, WT, bq, bk, bv,
                                                  qbuf, kbuf, vTb);
    flash_attn<<<2048, 128, 0, stream>>>(x, gamma, qbuf, kbuf, vTb, out);
}
